// SimpleONNXModel_27436251087120
// MI455X (gfx1250) — matrix operand order repaired and checked by reading, not compiled, not run on hardware
//
#include <hip/hip_runtime.h>
#include <hip/hip_bf16.h>

// ---------------------------------------------------------------------------
// DGCNN forward for MI455X (gfx1250, wave32, WMMA f32<-f16).
//
//   prep: x0 = [coords, intensity] -> (N,32) f16 (K-padded), xn = |x|^2
//   per EdgeConv l: dist chunks (WMMA, async-LDS double buffered) -> top-20
//                   -> GEMM [A;Bc] (WMMA) -> gather-max+BN+LReLU -> act, xn
//   concat(512) -> GEMM w5 (BN+LReLU) -> GEMM wf (BN) -> head GEMM -> d_out
//
// EdgeConv identity used:  max_k lrelu(bn(W.edge + b))
//   = lrelu(bn( max_k (W1 x)[idx[n,k]] + ((W2-W1) x)[n] + b ))   (bn scale>0)
// ---------------------------------------------------------------------------

typedef __attribute__((ext_vector_type(16))) _Float16 v16h;
typedef __attribute__((ext_vector_type(8)))  float    v8f;

#define NPTS 5000
#define NP   5120            // padded point count (multiple of 64)
#define KNN  20
#define CHUNK 1024           // distance-matrix row chunk

// Async global->LDS path (CDNA5); falls back to sync loads if not declared.
#if defined(__has_builtin)
#  if __has_builtin(__builtin_amdgcn_global_load_async_to_lds_b64) && \
      __has_builtin(__builtin_amdgcn_global_load_async_to_lds_b128) && \
      __has_builtin(__builtin_amdgcn_s_wait_asynccnt)
#    define CDNA5_ASYNC_LDS 1
#  endif
#endif
#ifndef CDNA5_ASYNC_LDS
#  define CDNA5_ASYNC_LDS 0
#endif

#if CDNA5_ASYNC_LDS
// Parameter types per clang-22 diagnostics: gcc-style int vectors, AS1 src / AS3 dst.
typedef int i32x2 __attribute__((__vector_size__(2 * sizeof(int))));
typedef int i32x4 __attribute__((__vector_size__(4 * sizeof(int))));
#define GLB2(p) ((__attribute__((address_space(1))) i32x2*)(p))
#define GLB4(p) ((__attribute__((address_space(1))) i32x4*)(p))
#define LDS2(p) ((__attribute__((address_space(3))) i32x2*)(p))
#define LDS4(p) ((__attribute__((address_space(3))) i32x4*)(p))
#endif

// ---- WMMA fragment builders (ISA 7.12.2 layouts, wave32) -------------------
// A 16x32 f16: lane L (m = L&15, half = L>>4), VGPR v: K = (v&4?16:0)+half*8+(v&3)*2
__device__ __forceinline__ v16h load_afrag(const _Float16* As /*32x32*/, int mL, int half) {
  v16h a;
#pragma unroll
  for (int e = 0; e < 16; ++e) {
    int v  = e >> 1;
    int kl = ((v & 4) ? 16 : 0) + half * 8 + ((v & 3) << 1) + (e & 1);
    a[e] = As[mL * 32 + kl];
  }
  return a;
}
// B 32x16 f16: lane L (n = L&15, half = L>>4), K = (e&8?16:0) + half*8 + (e&7)  (Bs is [col][k])
__device__ __forceinline__ v16h load_bfrag(const _Float16* Bs /*64x32*/, int cL, int half) {
  v16h b;
#pragma unroll
  for (int e = 0; e < 16; ++e) b[e] = Bs[cL * 32 + ((e >> 3) << 4) + half * 8 + (e & 7)];
  return b;
}

// ---- prep kernels ----------------------------------------------------------
__global__ __launch_bounds__(256) void prep_points(const float* __restrict__ coords,
                                                   const float* __restrict__ feats,
                                                   _Float16* __restrict__ act0,
                                                   float* __restrict__ xn) {
  int n = blockIdx.x * 256 + threadIdx.x;
  if (n >= NP) return;
  float v0 = 0.f, v1 = 0.f, v2 = 0.f, v3 = 0.f;
  if (n < NPTS) {
    v0 = coords[n * 3 + 0];
    v1 = coords[n * 3 + 1];
    v2 = coords[n * 3 + 2];
    v3 = feats[n * 4 + 3];          // intensity channel
  }
  xn[n] = v0 * v0 + v1 * v1 + v2 * v2 + v3 * v3;
  _Float16* row = act0 + (size_t)n * 32;
  row[0] = (_Float16)v0; row[1] = (_Float16)v1;
  row[2] = (_Float16)v2; row[3] = (_Float16)v3;
#pragma unroll
  for (int c = 4; c < 32; ++c) row[c] = (_Float16)0.f;
}

// Build Gt (Kp x 2*Cout) f16 from edge-conv weight W (Cout x 2*Cin):
//   Gt[i][o]        = W1[o][i]              (o <  Cout)
//   Gt[i][Cout+o']  = W2[o'][i] - W1[o'][i] (o >= Cout)
__global__ __launch_bounds__(256) void prep_wedge(const float* __restrict__ W,
                                                  _Float16* __restrict__ Gt,
                                                  int Cin, int Cout, int Kp) {
  int t = blockIdx.x * 256 + threadIdx.x;
  int M2 = 2 * Cout;
  if (t >= Kp * M2) return;
  int i = t / M2, o = t % M2;
  float v = 0.f;
  if (i < Cin) {
    if (o < Cout) v = W[(size_t)o * (2 * Cin) + i];
    else {
      int oo = o - Cout;
      v = W[(size_t)oo * (2 * Cin) + Cin + i] - W[(size_t)oo * (2 * Cin) + i];
    }
  }
  Gt[t] = (_Float16)v;
}

// Wt (Kc x Mp) f16 = transpose of W (Mout x Kc), zero-padded to Mp columns
__global__ __launch_bounds__(256) void prep_wplain(const float* __restrict__ W,
                                                   _Float16* __restrict__ Wt,
                                                   int Kc, int Mout, int Mp) {
  int t = blockIdx.x * 256 + threadIdx.x;
  if (t >= Kc * Mp) return;
  int i = t / Mp, o = t % Mp;
  Wt[t] = (_Float16)((o < Mout) ? W[(size_t)o * Kc + i] : 0.f);
}

// ---- fused distance-matrix kernel (WMMA) -----------------------------------
// D[r - rowBase][c] = xn[r] + xn[c] - 2 * <x_r, x_c>   for a CHUNK of rows.
// Async path: ping-pong LDS buffers; next tile's async global->LDS loads are
// issued right after the barrier so they overlap the current tile's WMMA.
__global__ __launch_bounds__(256) void dist_wmma(const _Float16* __restrict__ act,
                                                 const float* __restrict__ xn,
                                                 float* __restrict__ D,
                                                 int Kp, int rowBase) {
  const int t = threadIdx.x, lane = t & 31, wave = t >> 5;
  const int rowTile = wave >> 2, colTile = wave & 3;
  const int R0 = rowBase + blockIdx.y * 32;
  const int C0 = blockIdx.x * 64;
  const int half = lane >> 4;
  const int r = t >> 3, k4 = (t & 7) * 4;     // A-tile load slot (4 f16)
  const int p = t >> 2, k8 = (t & 3) * 8;     // B-tile load slot (8 f16)
  const _Float16* aSrc = act + (size_t)(R0 + r) * Kp + k4;
  const _Float16* bSrc = act + (size_t)(C0 + p) * Kp + k8;
  v8f acc = {};

#if CDNA5_ASYNC_LDS
  __shared__ _Float16 As[2][32 * 32];
  __shared__ _Float16 Bs[2][64 * 32];
  auto issue = [&](int buf, int kb) {
    __builtin_amdgcn_global_load_async_to_lds_b64(
        GLB2(aSrc + kb), LDS2(&As[buf][r * 32 + k4]), 0, 0);
    __builtin_amdgcn_global_load_async_to_lds_b128(
        GLB4(bSrc + kb), LDS4(&Bs[buf][p * 32 + k8]), 0, 0);
  };
  issue(0, 0);
  int cur = 0;
  for (int kb = 0; kb < Kp; kb += 32) {
    __builtin_amdgcn_s_wait_asynccnt(0);      // this wave's tile landed in LDS
    __syncthreads();                          // all waves' tiles visible
    if (kb + 32 < Kp) issue(cur ^ 1, kb + 32);
    v16h a = load_afrag(&As[cur][0], rowTile * 16 + (lane & 15), half);
    v16h b = load_bfrag(&Bs[cur][0], colTile * 16 + (lane & 15), half);
    acc = __builtin_amdgcn_wmma_f32_16x16x32_f16(false, a, false, b,
                                                 (short)0, acc, false, false);
    cur ^= 1;
  }
#else
  __shared__ _Float16 As[32 * 32];
  __shared__ _Float16 Bs[64 * 32];
  uint2 ra = *(const uint2*)aSrc;             // register double buffer
  uint4 rb = *(const uint4*)bSrc;
  for (int kb = 0; kb < Kp; kb += 32) {
    *(uint2*)&As[r * 32 + k4] = ra;
    *(uint4*)&Bs[p * 32 + k8] = rb;
    __syncthreads();
    if (kb + 32 < Kp) {
      ra = *(const uint2*)(aSrc + kb + 32);
      rb = *(const uint4*)(bSrc + kb + 32);
    }
    v16h a = load_afrag(As, rowTile * 16 + (lane & 15), half);
    v16h b = load_bfrag(Bs, colTile * 16 + (lane & 15), half);
    acc = __builtin_amdgcn_wmma_f32_16x16x32_f16(false, a, false, b,
                                                 (short)0, acc, false, false);
    __syncthreads();
  }
#endif

  const int col = C0 + colTile * 16 + (lane & 15);
  const float xnC = xn[col];
#pragma unroll
  for (int v = 0; v < 8; ++v) {
    int grow = R0 + rowTile * 16 + v + 8 * half;              // C/D layout
    float d = xn[grow] + xnC - 2.0f * acc[v];
    D[(size_t)(grow - rowBase) * NP + col] = d;
  }
}

// ---- top-20 selection (stable, matches top_k tie-breaking) -----------------
__global__ __launch_bounds__(128) void topk_kernel(const float* __restrict__ D,
                                                   int* __restrict__ idx,
                                                   int chunkRows, int rowBase) {
  int r = blockIdx.x * 128 + threadIdx.x;
  if (r >= chunkRows) return;
  const float* row = D + (size_t)r * NP;
  float bd[KNN]; int bi[KNN];
#pragma unroll
  for (int j = 0; j < KNN; ++j) { bd[j] = 3.4e38f; bi[j] = 0; }
  for (int c = 0; c < NPTS; ++c) {
    if ((c & 63) == 0) __builtin_prefetch(row + c + 256, 0, 0);
    float d = row[c];
    if (d < bd[KNN - 1]) {
      int j = KNN - 1;
      while (j > 0 && d < bd[j - 1]) { bd[j] = bd[j - 1]; bi[j] = bi[j - 1]; --j; }
      bd[j] = d; bi[j] = c;
    }
  }
  int gr = rowBase + r;
#pragma unroll
  for (int j = 0; j < KNN; ++j) idx[gr * KNN + j] = bi[j];
}

// ---- generic WMMA GEMM: out(NP x Mp) = act(NP x Kp) @ wt(Kp x Mp) ----------
// mode 0: raw f32                    mode 1: +bias, BN, LReLU, f32+f16
// mode 2: +bias, BN, f32+f16        mode 3: +bias, bounded (head -> d_out)
__global__ __launch_bounds__(256) void gemm_wmma(const _Float16* __restrict__ act,
                                                 const _Float16* __restrict__ wt,
                                                 float* __restrict__ out32,
                                                 _Float16* __restrict__ out16,
                                                 const float* __restrict__ bias,
                                                 const float* __restrict__ scale,
                                                 const float* __restrict__ shift,
                                                 int Kp, int Mp, int mode,
                                                 int Mreal, int outStride) {
  __shared__ _Float16 As[32 * 32];
  __shared__ _Float16 Bs[64 * 32];
  const int t = threadIdx.x, lane = t & 31, wave = t >> 5;
  const int rowTile = wave >> 2, colTile = wave & 3;
  const int R0 = blockIdx.y * 32;
  const int C0 = blockIdx.x * 64;
  const int half = lane >> 4;
  const int r = t >> 3, k4 = (t & 7) * 4;     // A-tile slot
  const int k = t >> 3, c0 = (t & 7) * 8;     // W-tile slot (transposed store)
  const _Float16* aSrc = act + (size_t)(R0 + r) * Kp + k4;
  v8f acc = {};
  uint2 ra = *(const uint2*)aSrc;             // register double buffer
  uint4 rw = *(const uint4*)&wt[(size_t)k * Mp + C0 + c0];
  for (int kb = 0; kb < Kp; kb += 32) {
    *(uint2*)&As[r * 32 + k4] = ra;
    { const _Float16* wp = (const _Float16*)&rw;
#pragma unroll
      for (int i = 0; i < 8; ++i) Bs[(c0 + i) * 32 + k] = wp[i]; }
    __syncthreads();
    if (kb + 32 < Kp) {
      ra = *(const uint2*)(aSrc + kb + 32);
      rw = *(const uint4*)&wt[(size_t)(kb + 32 + k) * Mp + C0 + c0];
    }
    v16h a = load_afrag(As, rowTile * 16 + (lane & 15), half);
    v16h b = load_bfrag(Bs, colTile * 16 + (lane & 15), half);
    acc = __builtin_amdgcn_wmma_f32_16x16x32_f16(false, a, false, b,
                                                 (short)0, acc, false, false);
    __syncthreads();
  }
  const int col = C0 + colTile * 16 + (lane & 15);
  float bi = 0.f, sc = 1.f, sh = 0.f;
  if (mode == 1 || mode == 2) { bi = bias[col]; sc = scale[col]; sh = shift[col]; }
  else if (mode == 3 && col < Mreal) bi = bias[col];
#pragma unroll
  for (int v = 0; v < 8; ++v) {
    int row = R0 + rowTile * 16 + v + 8 * half;
    float y = acc[v];
    if (mode == 0) {
      out32[(size_t)row * Mp + col] = y;
    } else if (mode == 3) {
      if (row < NPTS && col < Mreal) out32[(size_t)row * outStride + col] = y + bi;
    } else {
      y = sc * (y + bi) + sh;
      if (mode == 1) y = y > 0.f ? y : 0.2f * y;
      out32[(size_t)row * Mp + col] = y;
      out16[(size_t)row * Mp + col] = (_Float16)y;
    }
  }
}

// ---- gather-max + bias + BN + LReLU + next-layer norms ---------------------
// Y is (NP x 2*Cout): cols [0,Cout) = A = W1 x, cols [Cout,2Cout) = Bc.
__global__ __launch_bounds__(256) void gather_max_kernel(const float* __restrict__ Y,
                                                         const int* __restrict__ idx,
                                                         const float* __restrict__ bias,
                                                         const float* __restrict__ scale,
                                                         const float* __restrict__ shift,
                                                         _Float16* __restrict__ xc, int xcOff,
                                                         _Float16* __restrict__ actNext, int KpNext,
                                                         float* __restrict__ xnNext,
                                                         int Cout, int Mp) {
  __shared__ float red[256];
  const int n = blockIdx.x;
  const int o = threadIdx.x;           // blockDim.x == Cout (64/128/256)
  float y = 0.f;
  if (n < NPTS) {
    int jn[KNN];
#pragma unroll
    for (int kk = 0; kk < KNN; ++kk) jn[kk] = idx[n * KNN + kk];
    __builtin_prefetch(Y + (size_t)jn[0] * Mp + o, 0, 0);
    float m = -3.4e38f;
#pragma unroll 4
    for (int kk = 0; kk < KNN; ++kk) {
      if (kk + 1 < KNN) __builtin_prefetch(Y + (size_t)jn[kk + 1] * Mp + o, 0, 0);
      float v = Y[(size_t)jn[kk] * Mp + o];
      m = m > v ? m : v;
    }
    y = m + Y[(size_t)n * Mp + Cout + o] + bias[o];
    y = scale[o] * y + shift[o];
    y = y > 0.f ? y : 0.2f * y;
  }
  xc[(size_t)n * 512 + xcOff + o] = (_Float16)y;
  if (actNext) actNext[(size_t)n * KpNext + o] = (_Float16)y;
  if (xnNext) {
    red[o] = y * y;
    __syncthreads();
    for (int s = blockDim.x >> 1; s > 0; s >>= 1) {
      if (o < s) red[o] += red[o + s];
      __syncthreads();
    }
    if (o == 0) xnNext[n] = red[0];
  }
}

// ---------------------------------------------------------------------------
extern "C" void kernel_launch(void* const* d_in, const int* in_sizes, int n_in,
                              void* d_out, int out_size, void* d_ws, size_t ws_size,
                              hipStream_t stream) {
  (void)in_sizes; (void)n_in; (void)out_size; (void)ws_size;
  const float* coords = (const float*)d_in[0];
  const float* feats  = (const float*)d_in[1];
  const float* W[4]  = {(const float*)d_in[2],  (const float*)d_in[6],
                        (const float*)d_in[10], (const float*)d_in[14]};
  const float* Bb[4] = {(const float*)d_in[3],  (const float*)d_in[7],
                        (const float*)d_in[11], (const float*)d_in[15]};
  const float* Sc[4] = {(const float*)d_in[4],  (const float*)d_in[8],
                        (const float*)d_in[12], (const float*)d_in[16]};
  const float* Sh[4] = {(const float*)d_in[5],  (const float*)d_in[9],
                        (const float*)d_in[13], (const float*)d_in[17]};
  const float* w5 = (const float*)d_in[18]; const float* b5 = (const float*)d_in[19];
  const float* s5 = (const float*)d_in[20]; const float* sh5 = (const float*)d_in[21];
  const float* wf = (const float*)d_in[22]; const float* bf = (const float*)d_in[23];
  const float* sf = (const float*)d_in[24]; const float* shf = (const float*)d_in[25];
  const float* wh = (const float*)d_in[26]; const float* bh = (const float*)d_in[27];

  // ---- workspace layout ----
  char* ws = (char*)d_ws;
  size_t off = 0;
  auto alloc = [&](size_t bytes) -> char* {
    char* p = ws + off;
    off += (bytes + 255) & ~(size_t)255;
    return p;
  };
  float*     Dbuf  = (float*)    alloc((size_t)CHUNK * NP * 4);
  int*       knn   = (int*)      alloc((size_t)NPTS * KNN * 4);
  float*     xn    = (float*)    alloc((size_t)NP * 4);
  _Float16*  act0  = (_Float16*) alloc((size_t)NP * 32 * 2);
  _Float16*  act1  = (_Float16*) alloc((size_t)NP * 64 * 2);
  _Float16*  act2  = (_Float16*) alloc((size_t)NP * 64 * 2);
  _Float16*  act3  = (_Float16*) alloc((size_t)NP * 128 * 2);
  _Float16*  xc    = (_Float16*) alloc((size_t)NP * 512 * 2);
  _Float16*  y5h   = (_Float16*) alloc((size_t)NP * 1024 * 2);
  _Float16*  feath = (_Float16*) alloc((size_t)NP * 256 * 2);
  float*     Ybuf  = (float*)    alloc((size_t)NP * 1024 * 4);
  _Float16*  Gt0   = (_Float16*) alloc((size_t)32 * 128 * 2);
  _Float16*  Gt1   = (_Float16*) alloc((size_t)64 * 128 * 2);
  _Float16*  Gt2   = (_Float16*) alloc((size_t)64 * 256 * 2);
  _Float16*  Gt3   = (_Float16*) alloc((size_t)128 * 512 * 2);
  _Float16*  w5t   = (_Float16*) alloc((size_t)512 * 1024 * 2);
  _Float16*  wft   = (_Float16*) alloc((size_t)1024 * 256 * 2);
  _Float16*  wht   = (_Float16*) alloc((size_t)256 * 64 * 2);
  _Float16*  Gt[4] = {Gt0, Gt1, Gt2, Gt3};

  // ---- prep ----
  prep_points<<<NP / 256, 256, 0, stream>>>(coords, feats, act0, xn);
  prep_wedge<<<(32  * 128) / 256, 256, 0, stream>>>(W[0], Gt0,   4,  64,  32);
  prep_wedge<<<(64  * 128) / 256, 256, 0, stream>>>(W[1], Gt1,  64,  64,  64);
  prep_wedge<<<(64  * 256) / 256, 256, 0, stream>>>(W[2], Gt2,  64, 128,  64);
  prep_wedge<<<(128 * 512) / 256, 256, 0, stream>>>(W[3], Gt3, 128, 256, 128);
  prep_wplain<<<(512  * 1024) / 256, 256, 0, stream>>>(w5, w5t,  512, 1024, 1024);
  prep_wplain<<<(1024 * 256)  / 256, 256, 0, stream>>>(wf, wft, 1024,  256,  256);
  prep_wplain<<<(256  * 64)   / 256, 256, 0, stream>>>(wh, wht,  256,   20,   64);

  // ---- kNN over chunks ----
  auto knn_pass = [&](const _Float16* actL, int Kp) {
    for (int c = 0; c < 5; ++c) {
      int base = c * CHUNK;
      int rows = (base + CHUNK <= NPTS) ? CHUNK : (NPTS - base);
      dim3 g(NP / 64, (rows + 31) / 32);
      dist_wmma<<<g, 256, 0, stream>>>(actL, xn, Dbuf, Kp, base);
      topk_kernel<<<(rows + 127) / 128, 128, 0, stream>>>(Dbuf, knn, rows, base);
    }
  };
  // ---- one EdgeConv layer ----
  auto edge = [&](const _Float16* actL, int Kp, const _Float16* GtL, int Cout,
                  const float* b, const float* s, const float* sh,
                  _Float16* actNext, int KpNext, float* xnNext, int xcOff) {
    knn_pass(actL, Kp);
    int Mp = 2 * Cout;
    dim3 g(Mp / 64, NP / 32);
    gemm_wmma<<<g, 256, 0, stream>>>(actL, GtL, Ybuf, nullptr,
                                     nullptr, nullptr, nullptr,
                                     Kp, Mp, /*mode=*/0, 0, 0);
    gather_max_kernel<<<NP, Cout, 0, stream>>>(Ybuf, knn, b, s, sh,
                                               xc, xcOff, actNext, KpNext,
                                               xnNext, Cout, Mp);
  };

  edge(act0,  32, Gt[0],  64, Bb[0], Sc[0], Sh[0], act1,  64, xn,   0);
  edge(act1,  64, Gt[1],  64, Bb[1], Sc[1], Sh[1], act2,  64, xn,  64);
  edge(act2,  64, Gt[2], 128, Bb[2], Sc[2], Sh[2], act3, 128, xn, 128);
  edge(act3, 128, Gt[3], 256, Bb[3], Sc[3], Sh[3], nullptr, 0, nullptr, 256);

  // ---- MLP head ----
  gemm_wmma<<<dim3(1024 / 64, NP / 32), 256, 0, stream>>>(
      xc, w5t, Ybuf, y5h, b5, s5, sh5, 512, 1024, /*mode=*/1, 0, 0);
  gemm_wmma<<<dim3(256 / 64, NP / 32), 256, 0, stream>>>(
      y5h, wft, Ybuf, feath, bf, sf, shf, 1024, 256, /*mode=*/2, 0, 0);
  gemm_wmma<<<dim3(64 / 64, NP / 32), 256, 0, stream>>>(
      feath, wht, (float*)d_out, nullptr, bh, nullptr, nullptr,
      256, 64, /*mode=*/3, /*Mreal=*/20, /*outStride=*/20);
}
